// LinearAttentionTransformer_17051020165649
// MI455X (gfx1250) — matrix operand order repaired and checked by reading, not compiled, not run on hardware
//
#include <hip/hip_runtime.h>
#include <hip/hip_bf16.h>

// Problem constants (match reference)
#define D_IN  512
#define EMBED 256
#define NROW  8192

typedef __attribute__((ext_vector_type(16))) _Float16 v16h;
typedef __attribute__((ext_vector_type(8)))  _Float16 v8h;
typedef __attribute__((ext_vector_type(8)))  float    v8f;

// ---------------------------------------------------------------------------
// WMMA tile loaders (CDNA5 16x16x32 f16 layouts, ISA 7.12.2)
//
// A (16x32, MxK): lanes 0-15 hold M=0..15 with K = {0..7, 16..23};
//                 lanes 16-31 hold M=0..15 with K = {8..15, 24..31}.
// => per lane: two contiguous 8-half chunks at K offsets b and b+16, b=(lane/16)*8.
__device__ __forceinline__ v16h load_a_tile(const _Float16* base, int ld,
                                            int row0, int k0, int lane) {
  const int m = lane & 15;
  const int b = (lane >> 4) << 3;                       // 0 or 8
  const _Float16* p = base + (size_t)(row0 + m) * ld + k0 + b;
  v8h lo = *(const v8h*)(p);
  v8h hi = *(const v8h*)(p + 16);
  return __builtin_shufflevector(lo, hi, 0,1,2,3,4,5,6,7,8,9,10,11,12,13,14,15);
}

// B (32x16, KxN): lane%16 = N column; lanes 0-15 hold K = {0..7, 16..23}, lanes 16-31 K = {8..15, 24..31}.
// "base" is the operand stored column-major in K, i.e. row-major [Ncols][ld] so the
// K-values per lane are two contiguous 8-half chunks (two aligned 16-byte loads).
__device__ __forceinline__ v16h load_b_tile(const _Float16* base, int ld,
                                            int col0, int k0, int lane) {
  const int n  = lane & 15;
  const int kb = (lane >> 4) << 3;                      // 0 or 8
  const _Float16* p = base + (size_t)(col0 + n) * ld + k0 + kb;
  v8h lo = *(const v8h*)(p);
  v8h hi = *(const v8h*)(p + 16);
  return __builtin_shufflevector(lo, hi, 0,1,2,3,4,5,6,7,8,9,10,11,12,13,14,15);
}

// ---------------------------------------------------------------------------
// 0) f32 -> f16 conversion
__global__ void cvt_f16_kernel(const float* __restrict__ src,
                               _Float16* __restrict__ dst, int n) {
  int i = blockIdx.x * blockDim.x + threadIdx.x;
  if (i < n) dst[i] = (_Float16)src[i];
}

// ---------------------------------------------------------------------------
// 1) Projection GEMM:  O = X16 [NROW x D_IN] @ W16^T [D_IN x EMBED],
//    output stored TRANSPOSED as f16: Ot[EMBED][NROW]  (feeds S-GEMM as A and B).
//    One 16x16 output tile per wave, 8 waves/block. K-loop: 512/32 = 16 WMMAs.
__global__ void proj_kernel(const _Float16* __restrict__ X16,
                            const _Float16* __restrict__ W16,
                            _Float16* __restrict__ Ot) {
  const int wave = threadIdx.x >> 5;
  const int lane = threadIdx.x & 31;
  const int tile = blockIdx.x * 8 + wave;               // 8192 tiles: 512 row x 16 col
  const int row0 = (tile >> 4) << 4;                    // 0..8176
  const int e0   = (tile & 15) << 4;                    // 0..240

  v8f acc = {};
  for (int k0 = 0; k0 < D_IN; k0 += 32) {
    v16h a = load_a_tile(X16, D_IN, row0, k0, lane);
    v16h b = load_b_tile(W16, D_IN, e0,   k0, lane);    // W row-major == B K-contiguous
    acc = __builtin_amdgcn_wmma_f32_16x16x32_f16(false, a, false, b,
                                                 (short)0, acc, false, false);
  }
  // C/D layout: lanes 0-15 -> N=lane, M=r; lanes 16-31 -> N=lane-16, M=8+r.
  // Transposed store: Ot[e0+N][row0 + mbase + r] is contiguous in r -> one v8h store.
  const int n     = lane & 15;
  const int mbase = (lane >> 4) << 3;
  v8h out;
  #pragma unroll
  for (int r = 0; r < 8; ++r) out[r] = (_Float16)acc[r];
  *(v8h*)(Ot + (size_t)(e0 + n) * NROW + row0 + mbase) = out;
}

// ---------------------------------------------------------------------------
// 2) S = K^T V  [EMBED x EMBED], K-dim = NROW. Kt/Vt are [EMBED][NROW] f16.
//    256 output tiles, one per wave, 8 waves/block -> 32 blocks. 256 WMMAs/wave.
__global__ void s_kernel(const _Float16* __restrict__ Kt,
                         const _Float16* __restrict__ Vt,
                         float* __restrict__ S) {
  const int wave = threadIdx.x >> 5;
  const int lane = threadIdx.x & 31;
  const int tile = blockIdx.x * 8 + wave;               // 16x16 grid of 16x16 tiles
  const int m0 = (tile >> 4) << 4;
  const int e0 = (tile & 15) << 4;

  v8f acc = {};
  for (int k0 = 0; k0 < NROW; k0 += 32) {
    v16h a = load_a_tile(Kt, NROW, m0, k0, lane);       // A[m][k] = Kt[m][k] (= K^T)
    v16h b = load_b_tile(Vt, NROW, e0, k0, lane);       // B[k][e] = Vt[e][k]
    acc = __builtin_amdgcn_wmma_f32_16x16x32_f16(false, a, false, b,
                                                 (short)0, acc, false, false);
  }
  const int n     = lane & 15;
  const int mbase = (lane >> 4) << 3;
  #pragma unroll
  for (int r = 0; r < 8; ++r)
    S[(size_t)(m0 + mbase + r) * EMBED + e0 + n] = acc[r];
}

// ---------------------------------------------------------------------------
// 3) u = S @ head_w^T  [EMBED]   (one block of 256 threads)
__global__ void u_kernel(const float* __restrict__ S,
                         const float* __restrict__ hw,
                         float* __restrict__ u) {
  const int m = threadIdx.x;
  float s = 0.f;
  for (int e = 0; e < EMBED; ++e) s += S[(size_t)m * EMBED + e] * hw[e];
  u[m] = s;
}

// 4) w = W_Q^T @ u  [D_IN]   (one block of 512 threads; coalesced over d)
__global__ void w_kernel(const float* __restrict__ WQ,
                         const float* __restrict__ u,
                         float* __restrict__ w) {
  const int d = threadIdx.x;
  float s = 0.f;
  for (int m = 0; m < EMBED; ++m) s += WQ[(size_t)m * D_IN + d] * u[m];
  w[d] = s;
}

// 5) preds[n] = X[n,:] . w + b   (full f32; one wave per row, 8 rows/block)
__global__ void preds_kernel(const float* __restrict__ X,
                             const float* __restrict__ w,
                             const float* __restrict__ bias,
                             float* __restrict__ out) {
  const int wave = threadIdx.x >> 5;
  const int lane = threadIdx.x & 31;
  const int row  = blockIdx.x * 8 + wave;
  const float* xr = X + (size_t)row * D_IN;
  float s = 0.f;
  #pragma unroll 4
  for (int d = lane; d < D_IN; d += 32) s += xr[d] * w[d];
  #pragma unroll
  for (int off = 16; off > 0; off >>= 1) s += __shfl_down(s, off, 32);
  if (lane == 0) out[row] = s + bias[0];
}

// ---------------------------------------------------------------------------
extern "C" void kernel_launch(void* const* d_in, const int* in_sizes, int n_in,
                              void* d_out, int out_size, void* d_ws, size_t ws_size,
                              hipStream_t stream) {
  (void)in_sizes; (void)n_in; (void)out_size; (void)ws_size;

  const float* X   = (const float*)d_in[0];   // [8192, 512]
  const float* WQ  = (const float*)d_in[1];   // [256, 512]
  const float* WK  = (const float*)d_in[2];   // [256, 512]
  const float* WV  = (const float*)d_in[3];   // [256, 512]
  const float* hw  = (const float*)d_in[4];   // [1, 256]
  const float* hb  = (const float*)d_in[5];   // [1]
  float* preds     = (float*)d_out;           // [8192]

  // Workspace layout (bytes, 256-aligned)
  char* ws = (char*)d_ws;
  size_t off = 0;
  auto alloc = [&](size_t bytes) { void* p = ws + off; off = (off + bytes + 255) & ~(size_t)255; return p; };
  _Float16* X16  = (_Float16*)alloc((size_t)NROW * D_IN * 2);     // 8 MB
  _Float16* WK16 = (_Float16*)alloc((size_t)EMBED * D_IN * 2);    // 256 KB
  _Float16* WV16 = (_Float16*)alloc((size_t)EMBED * D_IN * 2);    // 256 KB
  _Float16* Kt   = (_Float16*)alloc((size_t)EMBED * NROW * 2);    // 4 MB
  _Float16* Vt   = (_Float16*)alloc((size_t)EMBED * NROW * 2);    // 4 MB
  float*    S    = (float*)   alloc((size_t)EMBED * EMBED * 4);   // 256 KB
  float*    u    = (float*)   alloc((size_t)EMBED * 4);
  float*    w    = (float*)   alloc((size_t)D_IN * 4);

  // 0) convert X, W_K, W_V to f16
  {
    int n = NROW * D_IN;
    cvt_f16_kernel<<<(n + 255) / 256, 256, 0, stream>>>(X, X16, n);
    n = EMBED * D_IN;
    cvt_f16_kernel<<<(n + 255) / 256, 256, 0, stream>>>(WK, WK16, n);
    cvt_f16_kernel<<<(n + 255) / 256, 256, 0, stream>>>(WV, WV16, n);
  }

  // 1) K^T and V^T projections via WMMA (8192 tiles / 8 waves = 1024 blocks each)
  proj_kernel<<<1024, 256, 0, stream>>>(X16, WK16, Kt);
  proj_kernel<<<1024, 256, 0, stream>>>(X16, WV16, Vt);

  // 2) S = K^T V via WMMA (256 tiles / 8 waves = 32 blocks)
  s_kernel<<<32, 256, 0, stream>>>(Kt, Vt, S);

  // 3-5) tiny epilogue chain
  u_kernel<<<1, EMBED, 0, stream>>>(S, hw, u);
  w_kernel<<<1, D_IN, 0, stream>>>(WQ, u, w);
  preds_kernel<<<NROW / 8, 256, 0, stream>>>(X, w, hb, preds);
}
